// DGCNNLayer_51402168599280
// MI455X (gfx1250) — matrix operand order repaired and checked by reading, not compiled, not run on hardware
//
#include <hip/hip_runtime.h>

#define N_PTS 16384
#define DIMS  64
#define KNN   16
#define CH    128   // hidden channels; also 2*DIMS edge feature width

typedef __attribute__((ext_vector_type(16))) __bf16        bf16x16;
typedef __attribute__((ext_vector_type(8)))  float         f32x8;
typedef __attribute__((ext_vector_type(4)))  unsigned int  u32x4;
typedef __attribute__((ext_vector_type(2)))  unsigned int  u32x2;

union Frag {
    bf16x16 v;
    u32x4   q[2];
};

static __device__ __forceinline__ unsigned short f32_to_bf16(float f) {
    unsigned u = __float_as_uint(f);
    unsigned r = (u + 0x7FFFu + ((u >> 16) & 1u)) >> 16;   // round-to-nearest-even
    return (unsigned short)r;
}
static __device__ __forceinline__ float bf16_to_f32(unsigned short h) {
    return __uint_as_float(((unsigned)h) << 16);
}

static __device__ __forceinline__ f32x8 wmma_bf16(const Frag& a, const Frag& b, f32x8 c) {
    return __builtin_amdgcn_wmma_f32_16x16x32_bf16(false, a.v, false, b.v,
                                                   (short)0, c, false, false);
}

// ---------------------------------------------------------------------------
// Kernel 1: per-row sum-of-squares + bf16 hi/lo split of x (for bf16x3 GEMM)
// ---------------------------------------------------------------------------
__global__ __launch_bounds__(256) void prep_rows(const float* __restrict__ x,
                                                 unsigned short* __restrict__ xhi,
                                                 unsigned short* __restrict__ xlo,
                                                 float* __restrict__ sq) {
    int i = blockIdx.x * blockDim.x + threadIdx.x;
    if (i >= N_PTS) return;
    const float4* row = (const float4*)(x + (size_t)i * DIMS);
    float s = 0.f;
#pragma unroll
    for (int t = 0; t < DIMS / 4; ++t) {
        float4 v = row[t];
        s += v.x * v.x + v.y * v.y + v.z * v.z + v.w * v.w;
        unsigned short h0 = f32_to_bf16(v.x), h1 = f32_to_bf16(v.y);
        unsigned short h2 = f32_to_bf16(v.z), h3 = f32_to_bf16(v.w);
        u32x2 ph, pl;
        ph.x = (unsigned)h0 | ((unsigned)h1 << 16);
        ph.y = (unsigned)h2 | ((unsigned)h3 << 16);
        pl.x = (unsigned)f32_to_bf16(v.x - bf16_to_f32(h0)) |
               ((unsigned)f32_to_bf16(v.y - bf16_to_f32(h1)) << 16);
        pl.y = (unsigned)f32_to_bf16(v.z - bf16_to_f32(h2)) |
               ((unsigned)f32_to_bf16(v.w - bf16_to_f32(h3)) << 16);
        *(u32x2*)(xhi + (size_t)i * DIMS + t * 4) = ph;
        *(u32x2*)(xlo + (size_t)i * DIMS + t * 4) = pl;
    }
    sq[i] = s;
}

// ---------------------------------------------------------------------------
// Kernel 2: transpose W1 (128x128) and W2 (128x128) to [col][k] bf16 layout
// ---------------------------------------------------------------------------
__global__ __launch_bounds__(256) void prep_weights(const float* __restrict__ W1,
                                                    const float* __restrict__ W2,
                                                    unsigned short* __restrict__ w1t,
                                                    unsigned short* __restrict__ w2t) {
    int t = blockIdx.x * blockDim.x + threadIdx.x;   // 0 .. 2*128*128-1
    int sel = t >> 14;
    int e   = t & 16383;
    int k = e >> 7, c = e & 127;
    if (!sel) w1t[c * CH + k] = f32_to_bf16(W1[k * CH + c]);
    else      w2t[c * CH + k] = f32_to_bf16(W2[k * CH + c]);
}

// ---------------------------------------------------------------------------
// Kernel 3: fused bf16x3 WMMA distance tiles + per-row top-16 (unordered)
// One wave per block, 32 rows per block (two 16-row WMMA M-tiles).
// ---------------------------------------------------------------------------
__global__ __launch_bounds__(32) void knn_kernel(const unsigned short* __restrict__ xhi,
                                                 const unsigned short* __restrict__ xlo,
                                                 const float* __restrict__ sq,
                                                 int* __restrict__ knn_idx) {
    __shared__ __align__(16) float dtile[32][16];
    __shared__ __align__(16) float hd[32][KNN];
    __shared__ __align__(16) int   hx[32][KNN];

    const int lane  = threadIdx.x;        // 0..31
    const int ibase = blockIdx.x * 32;
    const int mlow  = lane & 15;
    const int rhigh = (lane >= 16) ? 8 : 0;

    // --- A fragments for our two 16-row tiles (hi & lo, 2 K-chunks of 32) ---
    Frag Ahi[2][2], Alo[2][2];
#pragma unroll
    for (int t = 0; t < 2; ++t) {
        const size_t rbase = (size_t)(ibase + t * 16 + mlow) * DIMS;
#pragma unroll
        for (int q = 0; q < 2; ++q) {
            int kb = q * 32 + ((lane < 16) ? 0 : 8);   // A layout: K pairs, +16 for V4..7
            Ahi[t][q].q[0] = *(const u32x4*)(xhi + rbase + kb);
            Ahi[t][q].q[1] = *(const u32x4*)(xhi + rbase + kb + 16);
            Alo[t][q].q[0] = *(const u32x4*)(xlo + rbase + kb);
            Alo[t][q].q[1] = *(const u32x4*)(xlo + rbase + kb + 16);
        }
    }
    float sqi[2][8];
#pragma unroll
    for (int t = 0; t < 2; ++t)
#pragma unroll
        for (int r = 0; r < 8; ++r)
            sqi[t][r] = sq[ibase + t * 16 + rhigh + r];

    // --- init top-16 set ---
    float thr = 1e30f;
#pragma unroll
    for (int s = 0; s < KNN; ++s) { hd[lane][s] = 1e30f; hx[lane][s] = ibase + lane; }
    __syncthreads();

    for (int jb = 0; jb < N_PTS; jb += 16) {
        // B fragments for this 16-column tile (table-order K B striping)
        Frag Bh[2], Bl[2];
        const int    col   = jb + mlow;
        const size_t cbase = (size_t)col * DIMS;
#pragma unroll
        for (int q = 0; q < 2; ++q) {
            int kb = q * 32 + ((lane < 16) ? 0 : 8);
            Bh[q].q[0] = *(const u32x4*)(xhi + cbase + kb);
            Bh[q].q[1] = *(const u32x4*)(xhi + cbase + kb + 16);
            Bl[q].q[0] = *(const u32x4*)(xlo + cbase + kb);
            Bl[q].q[1] = *(const u32x4*)(xlo + cbase + kb + 16);
        }
        const float sqj = sq[col];

#pragma unroll
        for (int t = 0; t < 2; ++t) {
            f32x8 acc = {};
#pragma unroll
            for (int q = 0; q < 2; ++q) {
                acc = wmma_bf16(Ahi[t][q], Bh[q], acc);   // hi*hi
                acc = wmma_bf16(Ahi[t][q], Bl[q], acc);   // hi*lo
                acc = wmma_bf16(Alo[t][q], Bh[q], acc);   // lo*hi
            }
#pragma unroll
            for (int r = 0; r < 8; ++r)
                dtile[t * 16 + rhigh + r][mlow] = sqi[t][r] + sqj - 2.0f * acc[r];
        }
        __syncthreads();   // single-wave WG: lowers to S_NOP; orders LDS for compiler

        // --- merge 16 candidates of my row into top-16 set ---
#pragma unroll 4
        for (int c = 0; c < 16; ++c) {
            float d = dtile[lane][c];
            if (d < thr) {
                int worst = 0; float wmax = hd[lane][0];
#pragma unroll
                for (int s = 1; s < KNN; ++s) {
                    float v = hd[lane][s];
                    if (v > wmax) { wmax = v; worst = s; }
                }
                hd[lane][worst] = d;
                hx[lane][worst] = jb + c;
                wmax = hd[lane][0];
#pragma unroll
                for (int s = 1; s < KNN; ++s) { float v = hd[lane][s]; if (v > wmax) wmax = v; }
                thr = wmax;
            }
        }
        __syncthreads();
    }

#pragma unroll
    for (int s = 0; s < KNN; ++s)
        knn_idx[(size_t)(ibase + lane) * KNN + s] = hx[lane][s];
}

// ---------------------------------------------------------------------------
// Kernel 4: fused neighbor gather -> edge features -> 2x WMMA MLP -> mean.
// 4 waves/block, each wave processes 8 points; per point K=16 edges = one
// 16-row M-tile.  Weights stay hot in L2/WGP$ (64KB total).
// ---------------------------------------------------------------------------
__global__ __launch_bounds__(128) void mlp_kernel(const float* __restrict__ x,
                                                  const int* __restrict__ knn_idx,
                                                  const unsigned short* __restrict__ w1t,
                                                  const unsigned short* __restrict__ w2t,
                                                  const float* __restrict__ b1,
                                                  const float* __restrict__ b2,
                                                  float* __restrict__ out) {
    __shared__ __align__(16) unsigned short sE[4][KNN * CH];   // edge tiles per wave
    __shared__ __align__(16) unsigned short sH[4][KNN * CH];   // hidden tiles per wave

    const int wave = threadIdx.x >> 5;
    const int lane = threadIdx.x & 31;
    const int mlow = lane & 15;
    const int half = lane >> 4;            // 0: xi half / rows 0-7 ; 1: (xj-xi) half / rows 8-15
    unsigned short* E = &sE[wave][0];
    unsigned short* H = &sH[wave][0];

    for (int p = 0; p < 8; ++p) {
        const int i = (blockIdx.x * 4 + wave) * 8 + p;

        // ---- build 16x128 bf16 edge tile: row k = [ x_i , x_j - x_i ] ----
        const int j = knn_idx[(size_t)i * KNN + mlow];
        const float4* xi4 = (const float4*)(x + (size_t)i * DIMS);
        const float4* xj4 = (const float4*)(x + (size_t)j * DIMS);
#pragma unroll
        for (int s = 0; s < DIMS / 4; ++s) {
            float4 a = xi4[s];
            if (half) {
                float4 bq = xj4[s];
                a.x = bq.x - a.x; a.y = bq.y - a.y; a.z = bq.z - a.z; a.w = bq.w - a.w;
            }
            u32x2 pk;
            pk.x = (unsigned)f32_to_bf16(a.x) | ((unsigned)f32_to_bf16(a.y) << 16);
            pk.y = (unsigned)f32_to_bf16(a.z) | ((unsigned)f32_to_bf16(a.w) << 16);
            *(u32x2*)&E[mlow * CH + half * DIMS + s * 4] = pk;
        }

        // ---- layer 1: H = relu(E @ W1 + b1), 8 N-tiles x 4 K-chunks ----
        Frag A[4];
#pragma unroll
        for (int q = 0; q < 4; ++q) {
            int kb = q * 32 + (half ? 8 : 0);
            A[q].q[0] = *(const u32x4*)&E[mlow * CH + kb];
            A[q].q[1] = *(const u32x4*)&E[mlow * CH + kb + 16];
        }
#pragma unroll
        for (int nt = 0; nt < 8; ++nt) {
            f32x8 acc;
            const float bias = b1[nt * 16 + mlow];
#pragma unroll
            for (int r = 0; r < 8; ++r) acc[r] = bias;
            const size_t cb = (size_t)(nt * 16 + mlow) * CH;
#pragma unroll
            for (int q = 0; q < 4; ++q) {
                Frag B;
                int kb = q * 32 + (half ? 8 : 0);
                B.q[0] = *(const u32x4*)(w1t + cb + kb);
                B.q[1] = *(const u32x4*)(w1t + cb + kb + 16);
                acc = wmma_bf16(A[q], B, acc);
            }
#pragma unroll
            for (int r = 0; r < 8; ++r) {
                float v = acc[r] > 0.f ? acc[r] : 0.f;
                H[(r + half * 8) * CH + nt * 16 + mlow] = f32_to_bf16(v);
            }
        }

        // ---- layer 2 + mean over the 16 edges ----
        Frag A2[4];
#pragma unroll
        for (int q = 0; q < 4; ++q) {
            int kb = q * 32 + (half ? 8 : 0);
            A2[q].q[0] = *(const u32x4*)&H[mlow * CH + kb];
            A2[q].q[1] = *(const u32x4*)&H[mlow * CH + kb + 16];
        }
#pragma unroll
        for (int nt = 0; nt < 8; ++nt) {
            f32x8 acc;
            const float bias = b2[nt * 16 + mlow];
#pragma unroll
            for (int r = 0; r < 8; ++r) acc[r] = bias;
            const size_t cb = (size_t)(nt * 16 + mlow) * CH;
#pragma unroll
            for (int q = 0; q < 4; ++q) {
                Frag B;
                int kb = q * 32 + (half ? 8 : 0);
                B.q[0] = *(const u32x4*)(w2t + cb + kb);
                B.q[1] = *(const u32x4*)(w2t + cb + kb + 16);
                acc = wmma_bf16(A2[q], B, acc);
            }
            float s = 0.f;
#pragma unroll
            for (int r = 0; r < 8; ++r) { float v = acc[r]; s += (v > 0.f ? v : 0.f); }
            s += __shfl_xor(s, 16, 32);   // add the other half's 8 rows
            if (!half) out[(size_t)i * CH + nt * 16 + mlow] = s * (1.0f / KNN);
        }
    }
}

// ---------------------------------------------------------------------------
extern "C" void kernel_launch(void* const* d_in, const int* in_sizes, int n_in,
                              void* d_out, int out_size, void* d_ws, size_t ws_size,
                              hipStream_t stream) {
    const float* x  = (const float*)d_in[0];
    const float* W1 = (const float*)d_in[1];
    const float* b1 = (const float*)d_in[2];
    const float* W2 = (const float*)d_in[3];
    const float* b2 = (const float*)d_in[4];
    float* out = (float*)d_out;

    char* ws = (char*)d_ws;
    unsigned short* xhi = (unsigned short*)(ws);                    // 2 MB
    unsigned short* xlo = (unsigned short*)(ws + 2097152);          // 2 MB
    float*          sq  = (float*)         (ws + 4194304);          // 64 KB
    int*            knn = (int*)           (ws + 4259840);          // 1 MB
    unsigned short* w1t = (unsigned short*)(ws + 5308416);          // 32 KB
    unsigned short* w2t = (unsigned short*)(ws + 5341184);          // 32 KB

    prep_rows   <<<N_PTS / 256, 256, 0, stream>>>(x, xhi, xlo, sq);
    prep_weights<<<(2 * CH * CH) / 256, 256, 0, stream>>>(W1, W2, w1t, w2t);
    knn_kernel  <<<N_PTS / 32, 32, 0, stream>>>(xhi, xlo, sq, knn);
    mlp_kernel  <<<N_PTS / 32, 128, 0, stream>>>(x, knn, w1t, w2t, b1, b2, out);
}
